// BeliefAwareTrajAirNet_79405355368519
// MI455X (gfx1250) — matrix operand order repaired and checked by reading, not compiled, not run on hardware
//
#include <hip/hip_runtime.h>
#include <hip/hip_bf16.h>
#include <math.h>

// ---------------------------------------------------------------------------
// MI455X / gfx1250, wave32. All matrix math via v_wmma_f32_16x16x32_bf16.
// Each wave owns a 16 x (16*NT) C strip, 8 waves per 256-thread block.
//  * B operands PRE-PACKED once to bf16 fragment-native layout Bp[col][k]
//    (K zero-padded to x32): in-loop B fragment = two 16B-aligned b128 loads
//    bit-cast into v16bf. Zero conversion VALU, no tail clamps.
//  * A f32, converted u+=0x8000 + v_perm_b32 per pair (3 VALU / 2 elems);
//    per-lane base pointer with kbA folded -> all 16 A loads use constant
//    immediate offsets, one pointer bump per k-step.
//  * Loop body split into LOAD phase (all NT B fragments, one clause/wait)
//    then ISSUE phase (NT back-to-back v_wmma on distinct accumulators).
//  * M % 16 == 0 always -> no row guards; dead columns never stored ->
//    clamped addresses only; ragged K in one tail with A-side 0/1-mul mask.
// Fragment layouts per CDNA5 ISA 7.12.2 (A 16x32 / B 32x16 / C 8xVGPR f32).
// ---------------------------------------------------------------------------

#define WPB 8
#define TPB 256

typedef __attribute__((ext_vector_type(16))) __bf16    v16bf;
typedef __attribute__((ext_vector_type(8)))  float     v8f;
typedef __attribute__((ext_vector_type(8)))  unsigned  v8u;

__device__ __forceinline__ __bf16 f2bf(float f) {      // scalar (pack kernels)
    unsigned u = __builtin_bit_cast(unsigned, f);
    unsigned r = u + 0x7fffu + ((u >> 16) & 1u);
    unsigned short h = (unsigned short)(r >> 16);
    return __builtin_bit_cast(__bf16, h);
}

// two f32 -> packed bf16 pair: 2x add + 1x v_perm_b32
__device__ __forceinline__ unsigned pack2_bf16(float x, float y) {
    unsigned ux = __builtin_bit_cast(unsigned, x) + 0x8000u;
    unsigned uy = __builtin_bit_cast(unsigned, y) + 0x8000u;
    return __builtin_amdgcn_perm(uy, ux, 0x07060302u); // {uy[31:16], ux[31:16]}
}

// A-fragment pair base offset relative to (row base + kbA), slot v = 0..7:
// {0,2,4,6,16,18,20,22} -- compile-time constants
__device__ __forceinline__ constexpr int a_po(int v) {
    return (v < 4) ? 2 * v : 16 + 2 * (v - 4);
}
// K offset of pair v including lane-group base
__device__ __forceinline__ int a_koff(int v, int kbA) { return kbA + a_po(v); }
// K offset of element slot e (0..15)
__device__ __forceinline__ int a_off(int e, int kbA) {
    return a_koff(e >> 1, kbA) + (e & 1);
}

// 8 bf16 at p and 8 bf16 at p+16 -> v16bf via two b128 loads (ptr must be 16B aligned)
struct bfrag { uint4 lo, hi; };
__device__ __forceinline__ v16bf load_bfrag(const __bf16* p) {
    bfrag t; t.lo = *(const uint4*)p; t.hi = *(const uint4*)(p + 16);
    return __builtin_bit_cast(v16bf, t);
}

// ---------------------------------------------------------------------------
// One-shot pack kernels (off the hot path)
// ---------------------------------------------------------------------------
// B [K,N] row-major (ldb) -> Bp [N][Kp] bf16, zero-padded; batched on grid.y
__global__ void pack_bT(const float* __restrict__ B, __bf16* __restrict__ Bp,
                        int K, int N, int ldb, int Kp,
                        long long bB, long long bBp) {
    long long i = (long long)blockIdx.x * blockDim.x + threadIdx.x;
    if (i >= (long long)N * Kp) return;
    const float* Bh = B + (long long)blockIdx.y * bB;
    __bf16* Bph = Bp + (long long)blockIdx.y * bBp;
    int col = (int)(i / Kp), k = (int)(i % Kp);
    float v = k < K ? Bh[(size_t)k * ldb + col] : 0.f;
    Bph[i] = f2bf(v);
}
// W [rows][K] row-major -> Wp [rows][Kp] bf16, zero-padded (no transpose)
__global__ void pack_rows(const float* __restrict__ W, __bf16* __restrict__ Wp,
                          int rows, int K, int Kp) {
    long long i = (long long)blockIdx.x * blockDim.x + threadIdx.x;
    if (i >= (long long)rows * Kp) return;
    int r = (int)(i / Kp), k = (int)(i % Kp);
    Wp[i] = f2bf(k < K ? W[(size_t)r * K + k] : 0.f);
}

// ---------------------------------------------------------------------------
// Generic batched GEMM: C[b] = epi(A[b] @ B[b] + bias). M % 16 == 0 required.
// B pre-packed bf16 [N][Kp].  EPI: 0 = identity, 1 = relu.
// ---------------------------------------------------------------------------
template <int EPI, int NT>
__global__ void gemm_bf16_wmma(const float* __restrict__ A,
                               const __bf16* __restrict__ Bp,
                               const float* __restrict__ bias,
                               float* __restrict__ C,
                               int M, int N, int K,
                               int lda, int Kp, int ldc,
                               long long bA, long long bBp, long long bC) {
    const int lane = threadIdx.x & 31;
    const int wid  = threadIdx.x >> 5;
    const int tilesN = (N + 16 * NT - 1) / (16 * NT);
    const int tilesM = M >> 4;
    int tile = blockIdx.x * WPB + wid;
    if (tile >= tilesM * tilesN) return;
    const int b = blockIdx.y;
    A  += (long long)b * bA;
    Bp += (long long)b * bBp;
    C  += (long long)b * bC;
    const int tm = tile / tilesN, tn = tile % tilesN;
    const int row  = tm * 16 + (lane & 15);            // always < M
    const int col0 = tn * NT * 16 + (lane & 15);
    const int kbA = (lane >> 4) * 8;
    const int kbB = (lane >> 4) * 8;
    const float* ap = A + (size_t)row * lda + kbA;     // lane base, const offs
    const __bf16* bb[NT];                              // clamped col bases
#pragma unroll
    for (int nt = 0; nt < NT; ++nt) {
        int col = col0 + nt * 16;
        bb[nt] = Bp + (size_t)(col < N ? col : N - 1) * Kp + kbB;
    }
    v8f acc[NT];
#pragma unroll
    for (int nt = 0; nt < NT; ++nt) acc[nt] = v8f{};

    const int KF = K & ~31;
    int k0 = 0;
    for (; k0 < KF; k0 += 32, ap += 32) {              // ---- guard-free main
        v8u au;
#pragma unroll
        for (int v = 0; v < 8; ++v)
            au[v] = pack2_bf16(ap[a_po(v)], ap[a_po(v) + 1]);
        v16bf af = __builtin_bit_cast(v16bf, au);
        v16bf bfv[NT];                                 // load phase
#pragma unroll
        for (int nt = 0; nt < NT; ++nt) bfv[nt] = load_bfrag(bb[nt] + k0);
#pragma unroll
        for (int nt = 0; nt < NT; ++nt)                // issue phase
            acc[nt] = __builtin_amdgcn_wmma_f32_16x16x32_bf16(
                false, af, false, bfv[nt], (short)0, acc[nt], false, false);
    }
    if (k0 < K) {                                      // ---- single K tail
        v8u au;
#pragma unroll
        for (int v = 0; v < 8; ++v) {
            int kk  = k0 + kbA + a_po(v);
            int d0  = (kk     < K ? kk     : K - 1) - kbA - k0;
            int d1  = (kk + 1 < K ? kk + 1 : K - 1) - kbA - k0;
            float x0 = ap[d0] * ((kk     < K) ? 1.f : 0.f);
            float x1 = ap[d1] * ((kk + 1 < K) ? 1.f : 0.f);
            au[v] = pack2_bf16(x0, x1);                // A zeroed; Bp padded
        }
        v16bf af = __builtin_bit_cast(v16bf, au);
        v16bf bfv[NT];
#pragma unroll
        for (int nt = 0; nt < NT; ++nt) bfv[nt] = load_bfrag(bb[nt] + k0);
#pragma unroll
        for (int nt = 0; nt < NT; ++nt)
            acc[nt] = __builtin_amdgcn_wmma_f32_16x16x32_bf16(
                false, af, false, bfv[nt], (short)0, acc[nt], false, false);
    }
    const int rbase = tm * 16 + (lane >> 4) * 8;
#pragma unroll
    for (int nt = 0; nt < NT; ++nt) {
        const int col = col0 + nt * 16;
        if (col < N) {
            float bv = bias ? bias[col] : 0.f;
#pragma unroll
            for (int r = 0; r < 8; ++r) {
                float v = acc[nt][r] + bv;
                if (EPI == 1) v = fmaxf(v, 0.f);
                C[(size_t)(rbase + r) * ldc + col] = v;
            }
        }
    }
}

// ---------------------------------------------------------------------------
// Causal dilated conv as implicit-im2col WMMA GEMM, strength-reduced gather.
// Weights pre-packed bf16 [Cout][Kp]. Since k0 steps by 32, the tap k = kk&3
// is LOOP-INVARIANT per fragment slot: causal mask + ttc hoisted; address =
// base_e + (k0>>2)*strideC. FIRST: input [L, Cin, Nag], else [Nag, Cin, L].
// ---------------------------------------------------------------------------
template <int KK, bool FIRST, bool RELU, int NT>
__global__ void conv_causal_wmma(const float* __restrict__ X,
                                 const __bf16* __restrict__ Wp,
                                 const float* __restrict__ bias,
                                 float* __restrict__ O,
                                 int Nag, int Cin, int Cout, int L, int dil,
                                 int Kp) {
    const int lane = threadIdx.x & 31;
    const int wid  = threadIdx.x >> 5;
    const int M  = Nag * L;                            // multiple of 16
    const int KC = Cin * KK;
    const int KCF = KC & ~31;
    const int tilesN = (Cout + 16 * NT - 1) / (16 * NT);
    int tile = blockIdx.x * WPB + wid;
    if (tile >= (M >> 4) * tilesN) return;
    const int tm = tile / tilesN, tn = tile % tilesN;
    const int row  = tm * 16 + (lane & 15);            // always < M
    const int col0 = tn * NT * 16 + (lane & 15);
    const int kbA = (lane >> 4) * 8;
    const int kbB = (lane >> 4) * 8;
    const int n = row / L, t = row % L;                // computed once
    const int strideC = FIRST ? Nag : L;               // addr step per +4 in kk

    int   abase[16];                                   // loop-invariant bases
    float amask[16];                                   // causal 0/1 masks
#pragma unroll
    for (int e = 0; e < 16; ++e) {
        int off = a_off(e, kbA);
        int ci0 = (KK == 4) ? (off >> 2) : off;
        int k   = (KK == 4) ? (off & 3) : 0;
        int tt  = t - k * dil;
        int ttc = tt > 0 ? tt : 0;
        abase[e] = FIRST ? (ttc * Cin + ci0) * Nag + n
                         : (n * Cin + ci0) * L + ttc;
        amask[e] = tt >= 0 ? 1.f : 0.f;
    }
    const __bf16* wb[NT];
#pragma unroll
    for (int nt = 0; nt < NT; ++nt) {
        int col = col0 + nt * 16;
        wb[nt] = Wp + (size_t)(col < Cout ? col : Cout - 1) * Kp + kbB;
    }
    v8f acc[NT];
#pragma unroll
    for (int nt = 0; nt < NT; ++nt) acc[nt] = v8f{};

    int k0 = 0;
    for (; k0 < KCF; k0 += 32) {                       // ---- guard-free main
        const int koff = (k0 >> 2) * strideC;
        v8u au;
#pragma unroll
        for (int v = 0; v < 8; ++v)
            au[v] = pack2_bf16(X[abase[2 * v] + koff] * amask[2 * v],
                               X[abase[2 * v + 1] + koff] * amask[2 * v + 1]);
        v16bf af = __builtin_bit_cast(v16bf, au);
        v16bf bfv[NT];                                 // load phase
#pragma unroll
        for (int nt = 0; nt < NT; ++nt) bfv[nt] = load_bfrag(wb[nt] + k0);
#pragma unroll
        for (int nt = 0; nt < NT; ++nt)                // issue phase
            acc[nt] = __builtin_amdgcn_wmma_f32_16x16x32_bf16(
                false, af, false, bfv[nt], (short)0, acc[nt], false, false);
    }
    if (k0 < Kp) {                                     // ---- single K tail
        v8u au;
#pragma unroll
        for (int v = 0; v < 8; ++v) {
            float xs[2];
#pragma unroll
            for (int s = 0; s < 2; ++s) {
                int kk  = k0 + a_koff(v, kbA) + s;
                int ci  = (KK == 4) ? (kk >> 2) : kk;
                int k   = (KK == 4) ? (kk & 3) : 0;
                int cic = ci < Cin ? ci : Cin - 1;
                int tt  = t - k * dil;
                int ttc = tt > 0 ? tt : 0;
                float xv = FIRST ? X[(ttc * Cin + cic) * Nag + n]
                                 : X[(n * Cin + cic) * L + ttc];
                xs[s] = xv * ((ci < Cin && tt >= 0) ? 1.f : 0.f);
            }
            au[v] = pack2_bf16(xs[0], xs[1]);          // A zeroed; Wp padded
        }
        v16bf af = __builtin_bit_cast(v16bf, au);
        v16bf bfv[NT];
#pragma unroll
        for (int nt = 0; nt < NT; ++nt) bfv[nt] = load_bfrag(wb[nt] + k0);
#pragma unroll
        for (int nt = 0; nt < NT; ++nt)
            acc[nt] = __builtin_amdgcn_wmma_f32_16x16x32_bf16(
                false, af, false, bfv[nt], (short)0, acc[nt], false, false);
    }
    const int rbase = tm * 16 + (lane >> 4) * 8;
#pragma unroll
    for (int nt = 0; nt < NT; ++nt) {
        const int col = col0 + nt * 16;
        if (col < Cout) {
            float bv = bias[col];
#pragma unroll
            for (int r = 0; r < 8; ++r) {
                int m = rbase + r;
                int nn = m / L, tt = m % L;
                float v = acc[nt][r] + bv;
                if (RELU) v = fmaxf(v, 0.f);
                O[((size_t)nn * Cout + col) * L + tt] = v;
            }
        }
    }
}

// ---------------------------------------------------------------------------
// GAT: f_i = Wh_i . a[:nh], g_i = Wh_i . a[nh:]
// ---------------------------------------------------------------------------
__global__ void gat_fg(const float* __restrict__ Wh, const float* __restrict__ a,
                       float* __restrict__ f, float* __restrict__ g,
                       int N, int nh, long long bWh, long long ba) {
    int i = blockIdx.x * blockDim.x + threadIdx.x;
    int h = blockIdx.y;
    if (i >= N) return;
    const float* w  = Wh + (long long)h * bWh + (long long)i * nh;
    const float* ah = a + (long long)h * ba;
    float s0 = 0.f, s1 = 0.f;
    for (int c = 0; c < nh; ++c) { s0 += w[c] * ah[c]; s1 += w[c] * ah[nh + c]; }
    f[(long long)h * N + i] = s0;
    g[(long long)h * N + i] = s1;
}

// Pass 1 of softmax: per-row max and sum of exp over masked leaky(f_i+g_j).
__global__ void gat_rowstats(const float* __restrict__ f, const float* __restrict__ g,
                             const float* __restrict__ adj,
                             float* __restrict__ rmax, float* __restrict__ rsum, int N) {
    const int i = blockIdx.x, h = blockIdx.y;
    const float fi = f[(long long)h * N + i];
    const float* gh = g + (long long)h * N;
    const float* ar = adj + (long long)i * N;
    __shared__ float red[TPB];
    float m = -3.0e38f;
    for (int j = threadIdx.x; j < N; j += blockDim.x) {
        float e = fi + gh[j];
        e = fmaxf(e, 0.2f * e);                               // leaky relu
        e = ar[j] > 0.f ? e : -9.0e15f;                       // mask
        m = fmaxf(m, e);
    }
    red[threadIdx.x] = m; __syncthreads();
    for (int s = blockDim.x >> 1; s > 0; s >>= 1) {
        if (threadIdx.x < s) red[threadIdx.x] = fmaxf(red[threadIdx.x], red[threadIdx.x + s]);
        __syncthreads();
    }
    m = red[0]; __syncthreads();
    float sum = 0.f;
    for (int j = threadIdx.x; j < N; j += blockDim.x) {
        float e = fi + gh[j];
        e = fmaxf(e, 0.2f * e);
        e = ar[j] > 0.f ? e : -9.0e15f;
        sum += __expf(e - m);
    }
    red[threadIdx.x] = sum; __syncthreads();
    for (int s = blockDim.x >> 1; s > 0; s >>= 1) {
        if (threadIdx.x < s) red[threadIdx.x] += red[threadIdx.x + s];
        __syncthreads();
    }
    if (threadIdx.x == 0) {
        rmax[(long long)h * N + i] = m;
        rsum[(long long)h * N + i] = red[0];
    }
}

// Pass 2: out = elu(softmax(e) @ Wh). P generated inside the A fragment,
// once per k-step, reused over NT col tiles. Wh pre-packed bf16 [col][N].
// g/adj fetched with 16B-aligned float4 (global_load_b128); mask via multiply.
template <int NT>
__global__ void gat_attn_wmma(const float* __restrict__ f, const float* __restrict__ g,
                              const float* __restrict__ rmax, const float* __restrict__ rsum,
                              const float* __restrict__ adj, const __bf16* __restrict__ Bp,
                              float* __restrict__ Cc, int N, int nh,
                              long long bBp, long long bC, int ldc) {
    const int lane = threadIdx.x & 31;
    const int wid  = threadIdx.x >> 5;
    const int h = blockIdx.y;
    const int tilesN = (nh + 16 * NT - 1) / (16 * NT);
    int tile = blockIdx.x * WPB + wid;
    if (tile >= (N >> 4) * tilesN) return;
    const int tm = tile / tilesN, tn = tile % tilesN;
    const int row  = tm * 16 + (lane & 15);
    const int col0 = tn * NT * 16 + (lane & 15);
    const int kbA = (lane >> 4) * 8;
    const int kbB = (lane >> 4) * 8;
    const float* gh = g + (long long)h * N + kbA;      // lane base
    const float* ar = adj + (long long)row * N + kbA;
    const float fi  = f[(long long)h * N + row];
    const float mi  = rmax[(long long)h * N + row];
    const float inv = 1.f / rsum[(long long)h * N + row];
    const __bf16* bb[NT];
#pragma unroll
    for (int nt = 0; nt < NT; ++nt) {
        int col = col0 + nt * 16;
        bb[nt] = Bp + (long long)h * bBp + (size_t)(col < nh ? col : nh - 1) * N + kbB;
    }
    v8f acc[NT];
#pragma unroll
    for (int nt = 0; nt < NT; ++nt) acc[nt] = v8f{};
    for (int k0 = 0; k0 < N; k0 += 32, gh += 32, ar += 32) {   // K = N, no tail
        float4 gv[4] = {*(const float4*)gh,        *(const float4*)(gh + 4),
                        *(const float4*)(gh + 16), *(const float4*)(gh + 20)};
        float4 av[4] = {*(const float4*)ar,        *(const float4*)(ar + 4),
                        *(const float4*)(ar + 16), *(const float4*)(ar + 20)};
        const float* gs = (const float*)gv;
        const float* ms = (const float*)av;
        v8u au;
#pragma unroll
        for (int v = 0; v < 8; ++v) {
            float p[2];
#pragma unroll
            for (int s = 0; s < 2; ++s) {
                float ee = fi + gs[2 * v + s];
                ee = fmaxf(ee, 0.2f * ee);                    // leaky relu
                float pp = __expf(ee - mi) * inv;             // softmax prob
                p[s] = pp * (ms[2 * v + s] > 0.f ? 1.f : 0.f);
            }
            au[v] = pack2_bf16(p[0], p[1]);
        }
        v16bf af = __builtin_bit_cast(v16bf, au);
        v16bf bfv[NT];                                 // load phase
#pragma unroll
        for (int nt = 0; nt < NT; ++nt) bfv[nt] = load_bfrag(bb[nt] + k0);
#pragma unroll
        for (int nt = 0; nt < NT; ++nt)                // issue phase
            acc[nt] = __builtin_amdgcn_wmma_f32_16x16x32_bf16(
                false, af, false, bfv[nt], (short)0, acc[nt], false, false);
    }
    const int rbase = tm * 16 + (lane >> 4) * 8;
#pragma unroll
    for (int nt = 0; nt < NT; ++nt) {
        const int col = col0 + nt * 16;
        if (col < nh) {
#pragma unroll
            for (int r = 0; r < 8; ++r) {
                float v = acc[nt][r];
                v = v > 0.f ? v : (__expf(v) - 1.f);          // elu
                Cc[(long long)h * bC + (size_t)(rbase + r) * ldc + col] = v;
            }
        }
    }
}

// ---------------------------------------------------------------------------
// Glue kernels
// ---------------------------------------------------------------------------
__global__ void add_relu(const float* __restrict__ a, const float* __restrict__ b,
                         float* __restrict__ o, long long n) {
    long long i = (long long)blockIdx.x * blockDim.x + threadIdx.x;
    if (i >= n) return;
    o[i] = fmaxf(a[i] + b[i], 0.f);
}

__global__ void copy2d(const float* __restrict__ src, float* __restrict__ dst,
                       int rows, int cols, int lds, int ldd) {
    long long i = (long long)blockIdx.x * blockDim.x + threadIdx.x;
    if (i >= (long long)rows * cols) return;
    int r = (int)(i / cols), c = (int)(i % cols);
    dst[(long long)r * ldd + c] = src[(long long)r * lds + c];
}

// context conv(K=2, VALID) + linear + relu, written into spatial[:,132:139]
__global__ void ctx_kernel(const float* __restrict__ ctxin, const float* __restrict__ cw,
                           const float* __restrict__ cb, const float* __restrict__ lw,
                           const float* __restrict__ lb, float* __restrict__ spatial,
                           int N) {
    int n = blockIdx.x * blockDim.x + threadIdx.x;
    if (n >= N) return;
    float cc[10];
    for (int t = 0; t < 10; ++t) {
        float s = cb[0];
        for (int ch = 0; ch < 2; ++ch)
            for (int k = 0; k < 2; ++k)
                s += ctxin[((long long)(t + k) * 2 + ch) * N + n] * cw[ch * 2 + k];
        cc[t] = s;
    }
    for (int j = 0; j < 7; ++j) {
        float s = lb[j];
        for (int t = 0; t < 10; ++t) s += cc[t] * lw[t * 7 + j];
        spatial[(long long)n * 139 + 132 + j] = fmaxf(s, 0.f);
    }
}

__global__ void z_kernel(const float* __restrict__ mu, const float* __restrict__ lv,
                         const float* __restrict__ eps, float* __restrict__ z,
                         long long n) {
    long long i = (long long)blockIdx.x * blockDim.x + threadIdx.x;
    if (i >= n) return;
    z[i] = mu[i] + eps[i] * __expf(0.5f * lv[i]);
}

// acc = d.reshape(N,3,32)@dec_w + dec_b ; then double-integrate to absolute
__global__ void head_kernel(const float* __restrict__ d96, const float* __restrict__ dw,
                            const float* __restrict__ db, const float* __restrict__ x,
                            float* __restrict__ recon, int N) {
    int idx = blockIdx.x * blockDim.x + threadIdx.x;
    if (idx >= N * 3) return;
    int n = idx / 3, c = idx % 3;
    const float* dv = d96 + (long long)n * 96 + c * 32;
    float a[12];
    for (int t = 0; t < 12; ++t) {
        float s = db[t];
        for (int k = 0; k < 32; ++k) s += dv[k] * dw[k * 12 + t];
        a[t] = s;
    }
    float p1 = x[((long long)10 * 3 + c) * N + n];   // x[-1].T
    float p2 = x[((long long)9  * 3 + c) * N + n];   // x[-2].T
    for (int t = 0; t < 12; ++t) {
        float p = a[t] + 2.f * p1 - p2;
        recon[((long long)n * 3 + c) * 12 + t] = p;
        p2 = p1; p1 = p;
    }
}

// ---------------------------------------------------------------------------
// Host orchestration
// ---------------------------------------------------------------------------
static inline dim3 gemm_grid(int M, int N, int NT) {
    int tiles = ((M + 15) / 16) * ((N + 16 * NT - 1) / (16 * NT));
    return dim3((tiles + WPB - 1) / WPB);
}
static inline int kpad(int k) { return (k + 31) & ~31; }
static inline unsigned blk1d(long long n) { return (unsigned)((n + TPB - 1) / TPB); }

extern "C" void kernel_launch(void* const* d_in, const int* in_sizes, int n_in,
                              void* d_out, int out_size, void* d_ws, size_t ws_size,
                              hipStream_t stream) {
    (void)in_sizes; (void)n_in; (void)out_size; (void)ws_size;
    const int N = 2048;
    auto P = [&](int i) { return (const float*)d_in[i]; };
    // Input order: x, y, context, adj, eps, then params in insertion order:
    //  tcn_x blk0{w1,b1,w2,b2,wd,bd}=5..10 blk1{w1,b1,w2,b2}=11..14 blk2{..}=15..20
    //  tcn_y = 21..36 ; ctx_conv_w=37 ctx_conv_b=38 ctx_lin_w=39 ctx_lin_b=40
    //  gat W=41 a=42 Wout=43 aout=44 ; cvae_enc w0=45 b0=46 w1=47 b1=48
    //  mu_w=49 mu_b=50 lv_w=51 lv_b=52 ; cvae_dec w0=53 b0=54 w1=55 b1=56
    //  dec_w=57 dec_b=58
    const float* xin  = P(0);
    const float* ctxv = P(2);
    const float* adj  = P(3);
    const float* eps  = P(4);

    char* base = (char*)d_ws;
    size_t off = 0;
    auto allocB = [&](size_t bytes) {
        void* p = base + off; off += (bytes + 255) & ~(size_t)255; return p;
    };
    auto allocF = [&](size_t n) { return (float*)allocB(n * 4); };
    float* bufA   = allocF((size_t)N * 256 * 12);
    float* bufB   = allocF((size_t)N * 256 * 12);
    float* bufR   = allocF((size_t)N * 256 * 12);
    float* smA    = allocF((size_t)N * 12 * 12);
    float* smB    = allocF((size_t)N * 12 * 12);
    float* smC    = allocF((size_t)N * 12 * 12);
    float* encx   = allocF((size_t)N * 132);
    float* ency   = allocF((size_t)N * 144);
    float* spat   = allocF((size_t)N * 139);
    float* Wh     = allocF((size_t)16 * N * 64);
    float* fb     = allocF((size_t)16 * N);
    float* gb     = allocF((size_t)16 * N);
    float* rmax   = allocF((size_t)16 * N);
    float* rsum   = allocF((size_t)16 * N);
    float* hcat   = allocF((size_t)N * 1024);
    float* WhO    = allocF((size_t)N * 139);
    float* fO     = allocF(N); float* gO = allocF(N);
    float* mO     = allocF(N); float* sO = allocF(N);
    float* gout   = allocF((size_t)N * 139);
    float* Hx     = allocF((size_t)N * 278);
    float* encin  = allocF((size_t)N * 422);
    float* h1     = allocF((size_t)N * 128);
    float* h2     = allocF((size_t)N * 128);
    float* zb     = allocF((size_t)N * 128);
    float* decin  = allocF((size_t)N * 406);
    float* d1     = allocF((size_t)N * 128);
    float* d96    = allocF((size_t)N * 96);
    __bf16* wpack = (__bf16*)allocB((size_t)256 * 1024 * 2);     // conv weights
    __bf16* bpack = (__bf16*)allocB((size_t)512 * 1024 * 2);     // gemm weights
    __bf16* whp   = (__bf16*)allocB((size_t)16 * 64 * 2048 * 2); // packed Wh

    float* outp  = (float*)d_out;
    float* recon = outp;                 // [N,3,12]
    float* muOut = outp + (size_t)N * 36;
    float* lvOut = muOut + (size_t)N * 128;

    // ------------------ TCN (x and y encoders) ------------------
    auto run_tcn = [&](const float* in, int L, int p0, float* encOut) {
        const int M = N * L;
        long long nel = (long long)N * 256 * L;
        int kp;
        // block 0: 3 -> 256, dil 1 (+downsample)
        kp = kpad(3 * 4);
        pack_rows<<<blk1d((long long)256 * kp), TPB, 0, stream>>>(P(p0 + 0), wpack, 256, 12, kp);
        conv_causal_wmma<4, true, true, 4><<<gemm_grid(M, 256, 4), TPB, 0, stream>>>(
            in, wpack, P(p0 + 1), bufA, N, 3, 256, L, 1, kp);
        kp = kpad(256 * 4);
        pack_rows<<<blk1d((long long)256 * kp), TPB, 0, stream>>>(P(p0 + 2), wpack, 256, 1024, kp);
        conv_causal_wmma<4, false, true, 4><<<gemm_grid(M, 256, 4), TPB, 0, stream>>>(
            bufA, wpack, P(p0 + 3), bufB, N, 256, 256, L, 1, kp);
        kp = kpad(3);
        pack_rows<<<blk1d((long long)256 * kp), TPB, 0, stream>>>(P(p0 + 4), wpack, 256, 3, kp);
        conv_causal_wmma<1, true, false, 4><<<gemm_grid(M, 256, 4), TPB, 0, stream>>>(
            in, wpack, P(p0 + 5), bufR, N, 3, 256, L, 1, kp);
        add_relu<<<blk1d(nel), TPB, 0, stream>>>(bufB, bufR, bufA, nel);
        // block 1: 256 -> 256, dil 2 (identity residual)
        kp = kpad(256 * 4);
        pack_rows<<<blk1d((long long)256 * kp), TPB, 0, stream>>>(P(p0 + 6), wpack, 256, 1024, kp);
        conv_causal_wmma<4, false, true, 4><<<gemm_grid(M, 256, 4), TPB, 0, stream>>>(
            bufA, wpack, P(p0 + 7), bufB, N, 256, 256, L, 2, kp);
        pack_rows<<<blk1d((long long)256 * kp), TPB, 0, stream>>>(P(p0 + 8), wpack, 256, 1024, kp);
        conv_causal_wmma<4, false, true, 4><<<gemm_grid(M, 256, 4), TPB, 0, stream>>>(
            bufB, wpack, P(p0 + 9), bufR, N, 256, 256, L, 2, kp);
        add_relu<<<blk1d(nel), TPB, 0, stream>>>(bufR, bufA, bufB, nel);
        // block 2: 256 -> 12, dil 4 (+downsample)
        kp = kpad(256 * 4);
        pack_rows<<<blk1d((long long)12 * kp), TPB, 0, stream>>>(P(p0 + 10), wpack, 12, 1024, kp);
        conv_causal_wmma<4, false, true, 1><<<gemm_grid(M, 12, 1), TPB, 0, stream>>>(
            bufB, wpack, P(p0 + 11), smA, N, 256, 12, L, 4, kp);
        kp = kpad(12 * 4);
        pack_rows<<<blk1d((long long)12 * kp), TPB, 0, stream>>>(P(p0 + 12), wpack, 12, 48, kp);
        conv_causal_wmma<4, false, true, 1><<<gemm_grid(M, 12, 1), TPB, 0, stream>>>(
            smA, wpack, P(p0 + 13), smB, N, 12, 12, L, 4, kp);
        kp = kpad(256);
        pack_rows<<<blk1d((long long)12 * kp), TPB, 0, stream>>>(P(p0 + 14), wpack, 12, 256, kp);
        conv_causal_wmma<1, false, false, 1><<<gemm_grid(M, 12, 1), TPB, 0, stream>>>(
            bufB, wpack, P(p0 + 15), smC, N, 256, 12, L, 1, kp);
        long long ne2 = (long long)N * 12 * L;
        add_relu<<<blk1d(ne2), TPB, 0, stream>>>(smB, smC, encOut, ne2);
    };
    run_tcn(P(0), 11, 5, encx);    // enc_x: [N,132]
    run_tcn(P(1), 12, 21, ency);   // enc_y: [N,144]

    // ------------------ spatial = [enc_x | ctx] ------------------
    copy2d<<<blk1d((long long)N * 132), TPB, 0, stream>>>(encx, spat, N, 132, 132, 139);
    ctx_kernel<<<blk1d(N), TPB, 0, stream>>>(ctxv, P(37), P(38), P(39), P(40), spat, N);

    // ------------------ GAT, 16 heads batched ------------------
    {   // Wh[h] = spatial @ W[h] : pack W (16 heads) then batched GEMM
        int kp = kpad(139);
        pack_bT<<<dim3(blk1d((long long)64 * kp), 16), TPB, 0, stream>>>(
            P(41), bpack, 139, 64, 64, kp, (long long)139 * 64, (long long)64 * kp);
        gemm_bf16_wmma<0, 4><<<dim3(gemm_grid(N, 64, 4).x, 16), TPB, 0, stream>>>(
            spat, bpack, nullptr, Wh, N, 64, 139, 139, kp, 64,
            0LL, (long long)64 * kp, (long long)N * 64);
    }
    gat_fg<<<dim3(blk1d(N), 16), TPB, 0, stream>>>(Wh, P(42), fb, gb,
                                                   N, 64, (long long)N * 64, 128LL);
    gat_rowstats<<<dim3(N, 16), TPB, 0, stream>>>(fb, gb, adj, rmax, rsum, N);
    // pack Wh (B of attention) to bf16 [h][col][N]
    pack_bT<<<dim3(blk1d((long long)64 * N), 16), TPB, 0, stream>>>(
        Wh, whp, N, 64, 64, N, (long long)N * 64, (long long)64 * N);
    gat_attn_wmma<4><<<dim3(gemm_grid(N, 64, 4).x, 16), TPB, 0, stream>>>(
        fb, gb, rmax, rsum, adj, whp, hcat, N, 64, (long long)64 * N, 64LL, 1024);
    // output head over hcat [N,1024]
    {
        int kp = kpad(1024);
        pack_bT<<<blk1d((long long)139 * kp), TPB, 0, stream>>>(
            P(43), bpack, 1024, 139, 139, kp, 0LL, 0LL);
        gemm_bf16_wmma<0, 3><<<gemm_grid(N, 139, 3), TPB, 0, stream>>>(
            hcat, bpack, nullptr, WhO, N, 139, 1024, 1024, kp, 139, 0LL, 0LL, 0LL);
    }
    gat_fg<<<dim3(blk1d(N), 1), TPB, 0, stream>>>(WhO, P(44), fO, gO, N, 139, 0LL, 0LL);
    gat_rowstats<<<dim3(N, 1), TPB, 0, stream>>>(fO, gO, adj, mO, sO, N);
    pack_bT<<<blk1d((long long)139 * N), TPB, 0, stream>>>(
        WhO, bpack, N, 139, 139, N, 0LL, 0LL);
    gat_attn_wmma<3><<<dim3(gemm_grid(N, 139, 3).x, 1), TPB, 0, stream>>>(
        fO, gO, mO, sO, adj, bpack, gout, N, 139, 0LL, 0LL, 139);

    // ------------------ H_x = [spatial | gout] ------------------
    copy2d<<<blk1d((long long)N * 139), TPB, 0, stream>>>(spat, Hx, N, 139, 139, 278);
    copy2d<<<blk1d((long long)N * 139), TPB, 0, stream>>>(gout, Hx + 139, N, 139, 139, 278);

    // ------------------ CVAE encode ------------------
    copy2d<<<blk1d((long long)N * 144), TPB, 0, stream>>>(ency, encin, N, 144, 144, 422);
    copy2d<<<blk1d((long long)N * 278), TPB, 0, stream>>>(Hx, encin + 144, N, 278, 278, 422);
    {
        int kp = kpad(422);
        pack_bT<<<blk1d((long long)128 * kp), TPB, 0, stream>>>(P(45), bpack, 422, 128, 128, kp, 0LL, 0LL);
        gemm_bf16_wmma<1, 4><<<gemm_grid(N, 128, 4), TPB, 0, stream>>>(
            encin, bpack, P(46), h1, N, 128, 422, 422, kp, 128, 0LL, 0LL, 0LL);
    }
    pack_bT<<<blk1d((long long)128 * 128), TPB, 0, stream>>>(P(47), bpack, 128, 128, 128, 128, 0LL, 0LL);
    gemm_bf16_wmma<1, 4><<<gemm_grid(N, 128, 4), TPB, 0, stream>>>(
        h1, bpack, P(48), h2, N, 128, 128, 128, 128, 128, 0LL, 0LL, 0LL);
    pack_bT<<<blk1d((long long)128 * 128), TPB, 0, stream>>>(P(49), bpack, 128, 128, 128, 128, 0LL, 0LL);
    gemm_bf16_wmma<0, 4><<<gemm_grid(N, 128, 4), TPB, 0, stream>>>(
        h2, bpack, P(50), muOut, N, 128, 128, 128, 128, 128, 0LL, 0LL, 0LL);
    pack_bT<<<blk1d((long long)128 * 128), TPB, 0, stream>>>(P(51), bpack, 128, 128, 128, 128, 0LL, 0LL);
    gemm_bf16_wmma<0, 4><<<gemm_grid(N, 128, 4), TPB, 0, stream>>>(
        h2, bpack, P(52), lvOut, N, 128, 128, 128, 128, 128, 0LL, 0LL, 0LL);
    z_kernel<<<blk1d((long long)N * 128), TPB, 0, stream>>>(muOut, lvOut, eps, zb, (long long)N * 128);

    // ------------------ CVAE decode ------------------
    copy2d<<<blk1d((long long)N * 128), TPB, 0, stream>>>(zb, decin, N, 128, 128, 406);
    copy2d<<<blk1d((long long)N * 278), TPB, 0, stream>>>(Hx, decin + 128, N, 278, 278, 406);
    {
        int kp = kpad(406);
        pack_bT<<<blk1d((long long)128 * kp), TPB, 0, stream>>>(P(53), bpack, 406, 128, 128, kp, 0LL, 0LL);
        gemm_bf16_wmma<1, 4><<<gemm_grid(N, 128, 4), TPB, 0, stream>>>(
            decin, bpack, P(54), d1, N, 128, 406, 406, kp, 128, 0LL, 0LL, 0LL);
    }
    pack_bT<<<blk1d((long long)96 * 128), TPB, 0, stream>>>(P(55), bpack, 128, 96, 96, 128, 0LL, 0LL);
    gemm_bf16_wmma<0, 3><<<gemm_grid(N, 96, 3), TPB, 0, stream>>>(
        d1, bpack, P(56), d96, N, 96, 128, 128, 128, 96, 0LL, 0LL, 0LL);

    // ------------------ decoder head + double integration ------------------
    head_kernel<<<blk1d(N * 3), TPB, 0, stream>>>(d96, P(57), P(58), xin, recon, N);
}
